// ResidualBlock_downsample_12137577579237
// MI455X (gfx1250) — matrix operand order repaired and checked by reading, not compiled, not run on hardware
//
#include <hip/hip_runtime.h>

// ---------------------------------------------------------------------------
// CDNA5 (gfx1250) sparse-conv residual block.
// Matrix work is done with V_WMMA_F32_16X16X32_BF16 (f32 accumulate).
// ---------------------------------------------------------------------------

typedef __attribute__((ext_vector_type(16))) __bf16 v16bf;
typedef __attribute__((ext_vector_type(8)))  float  v8f;

#define BN_EPS 1e-4f
#define COUT   128   // Cout of every conv in this block

// round-to-nearest-even float -> bf16 bits
__device__ inline unsigned short f2bf(float f) {
  unsigned int u = __float_as_uint(f);
  unsigned int r = (u + 0x7FFFu + ((u >> 16) & 1u)) >> 16;
  return (unsigned short)r;
}

struct __attribute__((aligned(16))) U4 { unsigned int x, y, z, w; };

union Frag {        // 8 VGPRs = 16 bf16 values (one WMMA A or B fragment)
  U4 q[2];
  unsigned int u[8];
  v16bf v;
};

// ---------------------------------------------------------------------------
// BatchNorm: per-channel sum / sumsq (C is a power of two)
// ---------------------------------------------------------------------------
__global__ void bn_stats_kernel(const float* __restrict__ x, int nrows, int C,
                                float* __restrict__ stats) {
  int t = blockIdx.x * blockDim.x + threadIdx.x;
  int total = gridDim.x * blockDim.x;        // multiple of C by construction
  int c  = t & (C - 1);
  int r0 = t / C;
  int rs = total / C;
  float s = 0.f, s2 = 0.f;
  for (int r = r0; r < nrows; r += rs) {
    float v = x[(long long)r * C + c];
    s += v; s2 += v * v;
  }
  atomicAdd(&stats[c], s);
  atomicAdd(&stats[C + c], s2);
}

__global__ void bn_finalize_kernel(const float* __restrict__ stats,
                                   const float* __restrict__ gamma,
                                   const float* __restrict__ beta,
                                   int C, float nrows, float* __restrict__ prm) {
  int c = blockIdx.x * blockDim.x + threadIdx.x;
  if (c < C) {
    float mean = stats[c] / nrows;
    float var  = stats[C + c] / nrows - mean * mean;
    float sc   = gamma[c] * rsqrtf(var + BN_EPS);
    prm[c]     = sc;                 // scale
    prm[C + c] = beta[c] - mean * sc; // bias
  }
}

// apply BN + ReLU, emit bf16 activations
__global__ void bn_apply_kernel(const float* __restrict__ x,
                                const float* __restrict__ prm,
                                long long total, int C,
                                unsigned short* __restrict__ h) {
  long long i = (long long)blockIdx.x * blockDim.x + threadIdx.x;
  long long stride = (long long)gridDim.x * blockDim.x;
  for (; i < total; i += stride) {
    int c = (int)(i & (C - 1));
    float v = x[i] * prm[c] + prm[C + c];
    h[i] = f2bf(v > 0.f ? v : 0.f);
  }
}

// ---------------------------------------------------------------------------
// Pack fp32 weights [K][CIN][COUT] into a bf16 B-fragment lane layout (contiguous k per lane half):
// entry index = ((((k*KS + s)*8 + ntile)*32 + lane)*8 + j)   (one u32 per pair)
//   lane: N = ntile*16 + (lane&15), half = lane>>4
//   j   : K rows (s*32 + half*16 + 2j, +1) packed lo|hi
// A wave then loads its B fragment with two global_load_b128 per K-step.
// ---------------------------------------------------------------------------
__global__ void pack_w_kernel(const float* __restrict__ w, int K, int CIN,
                              unsigned int* __restrict__ wp) {
  int idx = blockIdx.x * blockDim.x + threadIdx.x;
  int KS = CIN >> 5;
  int total = K * KS * 8 * 32 * 8;
  if (idx >= total) return;
  int j    = idx & 7;
  int l    = (idx >> 3) & 31;
  int n    = (idx >> 8) & 7;
  int rest = idx >> 11;           // k*KS + s
  int s = rest % KS;
  int k = rest / KS;
  int half = l >> 4;
  int col  = n * 16 + (l & 15);
  int Kr   = s * 32 + half * 16 + 2 * j;
  const float* wk = w + (long long)k * CIN * COUT;
  unsigned int lo = f2bf(wk[(long long)Kr * COUT + col]);
  unsigned int hi = f2bf(wk[(long long)(Kr + 1) * COUT + col]);
  wp[idx] = lo | (hi << 16);
}

// ---------------------------------------------------------------------------
// Gather -> WMMA -> scatter-add sparse convolution.
// Block: 256 threads (8 wave32), wave w owns Cout tile [16w, 16w+16).
// Block handles 64 rulebook pairs of one kernel offset k (4 M-subtiles of 16).
// A fragment (16x32 bf16): lane half selects K run {16h..16h+15}
// which is one contiguous channel run -> two aligned b128 loads per K-step.
// ---------------------------------------------------------------------------
template <int CIN>
__global__ void __launch_bounds__(256)
sparse_conv_wmma_kernel(const unsigned short* __restrict__ h,   // [N_in][CIN] bf16
                        const unsigned int*  __restrict__ wp,   // packed weights
                        const int* __restrict__ in_idx,         // [K][P]
                        const int* __restrict__ out_idx,        // [K][P]
                        int P, float* __restrict__ out) {       // [n_out][COUT]
  constexpr int KS = CIN / 32;
  const int lane = threadIdx.x & 31;
  const int wave = threadIdx.x >> 5;        // n-tile 0..7
  const int half = lane >> 4;
  const int mloc = lane & 15;
  const int k    = blockIdx.y;
  const int p0   = blockIdx.x * 64;
  const long long kP = (long long)k * P;

  // B fragments for this wave's Cout tile (reused across 64 pairs)
  Frag b[KS];
#pragma unroll
  for (int s = 0; s < KS; ++s) {
    const U4* q = (const U4*)(wp +
        ((((long long)(k * KS + s)) * 8 + wave) * 32 + lane) * 8);
    b[s].q[0] = q[0];
    b[s].q[1] = q[1];
  }

  for (int sub = 0; sub < 4; ++sub) {
    int pbase = p0 + sub * 16;
    int p  = pbase + mloc;
    int pc = p < P ? p : P - 1;             // clamp (keeps EXEC uniform)
    int row = in_idx[kP + pc];
    const unsigned short* hr = h + (long long)row * CIN;

    Frag a[KS];
#pragma unroll
    for (int s = 0; s < KS; ++s) {
      int off = s * 32 + half * 16;         // even element offset, 16B aligned
      a[s].q[0] = *(const U4*)(hr + off);
      a[s].q[1] = *(const U4*)(hr + off + 8);
    }

    v8f acc = {};
#pragma unroll
    for (int s = 0; s < KS; ++s)
      acc = __builtin_amdgcn_wmma_f32_16x16x32_bf16(
          false, a[s].v, false, b[s].v, (short)0, acc, false, false);

    // C/D layout: VGPR r -> (M = r + 8*half, N = lane&15)
    int ncol = wave * 16 + mloc;
#pragma unroll
    for (int r = 0; r < 8; ++r) {
      int m  = r + half * 8;
      int pp = pbase + m;
      if (pp < P) {
        int orow = out_idx[kP + pp];
        atomicAdd(&out[(long long)orow * COUT + ncol], acc[r]);
      }
    }
  }
}

// ---------------------------------------------------------------------------
// Host launcher
// ---------------------------------------------------------------------------
extern "C" void kernel_launch(void* const* d_in, const int* in_sizes, int n_in,
                              void* d_out, int out_size, void* d_ws, size_t ws_size,
                              hipStream_t stream) {
  const float* x      = (const float*)d_in[0];
  const float* gamma1 = (const float*)d_in[1];
  const float* beta1  = (const float*)d_in[2];
  const float* w1     = (const float*)d_in[3];
  const float* w2     = (const float*)d_in[4];
  const float* gamma2 = (const float*)d_in[5];
  const float* beta2  = (const float*)d_in[6];
  const float* wsub   = (const float*)d_in[7];
  const int*   rb_in  = (const int*)d_in[8];
  const int*   rb_out = (const int*)d_in[9];
  const int*   sub_in = (const int*)d_in[10];
  const int*   sub_out= (const int*)d_in[11];

  const int C   = in_sizes[1];                 // 64
  const int C2  = in_sizes[5];                 // 128 (== COUT)
  const int N_in = in_sizes[0] / C;
  const int K    = in_sizes[3] / (C * C2);
  const int P1   = in_sizes[8] / K;
  const int P2   = in_sizes[10] / K;
  const int n_out = out_size / C2;

  // --- workspace carve-out (256B aligned) ---
  char* ws = (char*)d_ws;
  size_t off = 0;
  auto carve = [&](size_t bytes) {
    void* p = ws + off;
    off = (off + bytes + 255) & ~(size_t)255;
    return p;
  };
  float* stats1 = (float*)carve(2 * C  * sizeof(float));
  float* prm1   = (float*)carve(2 * C  * sizeof(float));
  float* stats2 = (float*)carve(2 * C2 * sizeof(float));
  float* prm2   = (float*)carve(2 * C2 * sizeof(float));
  unsigned int* w1p  = (unsigned int*)carve((size_t)K * C  * C2 * 2);
  unsigned int* w2p  = (unsigned int*)carve((size_t)K * C  * C2 * 2);
  unsigned int* wsp  = (unsigned int*)carve((size_t)K * C2 * C2 * 2);
  unsigned short* hbf  = (unsigned short*)carve((size_t)N_in * C  * 2);
  float*          y1   = (float*)carve((size_t)n_out * C2 * 4);
  unsigned short* y1bf = (unsigned short*)carve((size_t)n_out * C2 * 2);
  (void)ws_size; (void)n_in;

  // --- zero accumulators (graph-capture safe) ---
  hipMemsetAsync(stats1, 0, 2 * C  * sizeof(float), stream);
  hipMemsetAsync(stats2, 0, 2 * C2 * sizeof(float), stream);
  hipMemsetAsync(y1,     0, (size_t)n_out * C2 * sizeof(float), stream);
  hipMemsetAsync(d_out,  0, (size_t)out_size * sizeof(float), stream);

  // --- weight packing (bf16, WMMA B-fragment layout) ---
  {
    int tot1 = K * (C  / 32) * 8 * 32 * 8;
    int tot2 = K * (C2 / 32) * 8 * 32 * 8;
    pack_w_kernel<<<(tot1 + 255) / 256, 256, 0, stream>>>(w1,   K, C,  w1p);
    pack_w_kernel<<<(tot1 + 255) / 256, 256, 0, stream>>>(w2,   K, C,  w2p);
    pack_w_kernel<<<(tot2 + 255) / 256, 256, 0, stream>>>(wsub, K, C2, wsp);
  }

  // --- bnr1: stats -> params -> apply (fp32 -> bf16) ---
  bn_stats_kernel<<<512, 256, 0, stream>>>(x, N_in, C, stats1);
  bn_finalize_kernel<<<1, 256, 0, stream>>>(stats1, gamma1, beta1, C,
                                            (float)N_in, prm1);
  bn_apply_kernel<<<2048, 256, 0, stream>>>(x, prm1, (long long)N_in * C, C, hbf);

  // --- conv1 (-> y1) and conv2 shortcut (-> d_out) ---
  {
    dim3 grid((P1 + 63) / 64, K);
    sparse_conv_wmma_kernel<64><<<grid, 256, 0, stream>>>(
        hbf, w1p, rb_in, rb_out, P1, y1);
    sparse_conv_wmma_kernel<64><<<grid, 256, 0, stream>>>(
        hbf, w2p, rb_in, rb_out, P1, (float*)d_out);
  }

  // --- bnr2 over y1 -> bf16 ---
  bn_stats_kernel<<<512, 256, 0, stream>>>(y1, n_out, C2, stats2);
  bn_finalize_kernel<<<1, 256, 0, stream>>>(stats2, gamma2, beta2, C2,
                                            (float)n_out, prm2);
  bn_apply_kernel<<<2048, 256, 0, stream>>>(y1, prm2, (long long)n_out * C2,
                                            C2, y1bf);

  // --- submanifold conv (accumulates into d_out => y1_sub + y2) ---
  {
    dim3 grid((P2 + 63) / 64, K);
    sparse_conv_wmma_kernel<128><<<grid, 256, 0, stream>>>(
        y1bf, wsp, sub_in, sub_out, P2, (float*)d_out);
  }
}
